// DualGATv2_EarlyFusion_15418932592802
// MI455X (gfx1250) — matrix operand order repaired and checked by reading, not compiled, not run on hardware
//
#include <hip/hip_runtime.h>
#include <hip/hip_bf16.h>

// ---------------------------------------------------------------------------
// DualGATv2 early-fusion forward for gfx1250 (MI455X), wave32 + WMMA bf16.
// ---------------------------------------------------------------------------

typedef __bf16 v16bf __attribute__((ext_vector_type(16)));
typedef float  v8f   __attribute__((ext_vector_type(8)));

#define NNODES 46080
#define NEDGES 1474560
#define NGRAPH 128
#define NODESPG 360
#define HC 128
#define STAT_ROWS 256
#define STAT_CHUNKS 180   // ceil(46080/256)

// ---- order-preserving float <-> uint encoding (for atomicMax on floats) ----
__device__ __forceinline__ unsigned fenc(float f) {
    unsigned u = __float_as_uint(f);
    return (u & 0x80000000u) ? ~u : (u | 0x80000000u);
}
__device__ __forceinline__ float fdec(unsigned u) {
    unsigned b = (u & 0x80000000u) ? (u & 0x7FFFFFFFu) : ~u;
    return __uint_as_float(b);
}

// ---------------------------------------------------------------------------
// Node embedding: out[n,j] = relu(x[n,0]*W[0,j] + x[n,1]*W[1,j] + b[j]), j<64
// ---------------------------------------------------------------------------
__global__ void embed_kernel(const float* __restrict__ x,
                             const float* __restrict__ W,
                             const float* __restrict__ b,
                             float* __restrict__ out, int n_) {
    int t = blockIdx.x * blockDim.x + threadIdx.x;
    if (t >= n_ * 64) return;
    int nn = t >> 6, j = t & 63;
    float v = x[nn * 2] * W[j] + x[nn * 2 + 1] * W[64 + j] + b[j];
    out[t] = v > 0.f ? v : 0.f;
}

// ---------------------------------------------------------------------------
// Weight pre-pack: convert W [din,128] f32 (row-major) into bf16 WMMA
// B-fragment order, so the GEMM can load B as one coalesced 32B vector/lane.
// Fragment layout (ISA 7.12.2, 32x16 bf16 B): lane l -> n = l&15, g = l>>4;
// element j (0..15): K = s*32 + g*8 + (j&7) + (j>>3)*16.  One thread per (which,s,tn,lane).
// ---------------------------------------------------------------------------
__global__ void pack_wmma_b(const float* __restrict__ Wl, const float* __restrict__ Wr,
                            unsigned short* __restrict__ pl, unsigned short* __restrict__ pr,
                            int ksteps) {
    int t = blockIdx.x * blockDim.x + threadIdx.x;
    int total = ksteps * 8 * 32;
    if (t >= 2 * total) return;
    int which = t / total;
    int r = t - which * total;
    int lane = r & 31; r >>= 5;
    int tn = r & 7;
    int s  = r >> 3;
    const float* W = which ? Wr : Wl;
    unsigned short* P = which ? pr : pl;
    int g = lane >> 4, n = lane & 15;
    const float* col = W + (size_t)(s * 32 + g * 8) * HC + tn * 16 + n;
    unsigned short* o = P + ((size_t)((s * 8 + tn) * 32 + lane) * 16);
#pragma unroll
    for (int j = 0; j < 16; ++j) {
        __bf16 v = (__bf16)col[((j & 7) + (j >> 3) * 16) * HC];
        o[j] = __builtin_bit_cast(unsigned short, v);
    }
}

// ---------------------------------------------------------------------------
// WMMA GEMM: one wave per 16-row strip, computing all 8 column tiles of
// out[nrows,128] = in[nrows,DIN] @ W + bias.  A fragment converted once per
// k-step and reused by 8 back-to-back wmmas; B read pre-packed (coalesced).
// which=0 -> (pl,bl,xl), which=1 -> (pr,br,xr).
// ---------------------------------------------------------------------------
template <int DIN>
__global__ void gemm_strip_wmma(const float* __restrict__ in,
                                const unsigned short* __restrict__ pl,
                                const unsigned short* __restrict__ pr,
                                const float* __restrict__ bl, const float* __restrict__ br,
                                float* __restrict__ xl, float* __restrict__ xr,
                                int nrows) {
    int wave = (blockIdx.x * blockDim.x + threadIdx.x) >> 5;
    int lane = threadIdx.x & 31;
    int rowTiles = nrows >> 4;
    int which = wave / rowTiles;
    if (which >= 2) return;
    int tm = wave - which * rowTiles;
    const unsigned short* P = which ? pr : pl;
    const float* bias = which ? br : bl;
    float*       out  = which ? xr : xl;

    int row0 = tm * 16;
    int g = lane >> 4;      // K-half select
    int m = lane & 15;      // A row / D column within tile

    v8f acc[8];
#pragma unroll
    for (int tn = 0; tn < 8; ++tn) acc[tn] = (v8f){};

    const int KS = DIN / 32;
#pragma unroll
    for (int s = 0; s < KS; ++s) {
        // ---- A fragment: 16x32 bf16 (ISA 7.12.2 layout) ----
        const float* arow = in + (size_t)(row0 + m) * DIN + s * 32 + g * 8;
        v16bf a;
#pragma unroll
        for (int r = 0; r < 4; ++r) {
            a[2 * r]         = (__bf16)arow[2 * r];          // K = g*8 + 2r
            a[2 * r + 1]     = (__bf16)arow[2 * r + 1];
            a[8 + 2 * r]     = (__bf16)arow[16 + 2 * r];     // K = 16 + g*8 + 2r
            a[8 + 2 * r + 1] = (__bf16)arow[16 + 2 * r + 1];
        }
        // ---- 8 column tiles against this A fragment ----
#pragma unroll
        for (int tn = 0; tn < 8; ++tn) {
            const v16bf* bp =
                (const v16bf*)(P + ((size_t)((s * 8 + tn) * 32 + lane) * 16));
            v16bf b = *bp;
            acc[tn] = __builtin_amdgcn_wmma_f32_16x16x32_bf16(
                false, a, false, b, (short)0, acc[tn], false, false);
        }
    }
    // ---- D: VGPR r -> row (g*8 + r), column (lane&15) ----
    int mb = g * 8;
#pragma unroll
    for (int tn = 0; tn < 8; ++tn) {
        float bj = bias[tn * 16 + m];
#pragma unroll
        for (int r = 0; r < 8; ++r)
            out[(size_t)(row0 + mb + r) * HC + tn * 16 + m] = acc[tn][r] + bj;
    }
}

// ---------------------------------------------------------------------------
// Edge pass 1: logit[e,h] = sum_c att[h,c]*lrelu(xl[src]+xr[dst]+ea*We)
// and running segment max into mxenc[dst,h] (encoded-uint atomicMax).
// ---------------------------------------------------------------------------
__global__ void gat_edge_logits(const float* __restrict__ xl, const float* __restrict__ xr,
                                const int* __restrict__ ei, const float* __restrict__ ea,
                                const float* __restrict__ We, const float* __restrict__ att,
                                float* __restrict__ logit, unsigned* __restrict__ mxenc,
                                int E_) {
    int t = blockIdx.x * blockDim.x + threadIdx.x;
    if (t >= E_ * 8) return;
    int e = t >> 3, h = t & 7;
    int src = ei[e], dst = ei[E_ + e];
    float eav = ea[e];
    const float4* pl = (const float4*)(xl + (size_t)src * HC + h * 16);
    const float4* pr = (const float4*)(xr + (size_t)dst * HC + h * 16);
    const float4* pw = (const float4*)(We + h * 16);
    const float4* pa = (const float4*)(att + h * 16);
    float lg = 0.f;
#pragma unroll
    for (int q = 0; q < 4; ++q) {
        float4 A = pl[q], Bv = pr[q], Wv = pw[q], Tv = pa[q];
        float m0 = A.x + Bv.x + eav * Wv.x; m0 = m0 > 0.f ? m0 : 0.2f * m0;
        float m1 = A.y + Bv.y + eav * Wv.y; m1 = m1 > 0.f ? m1 : 0.2f * m1;
        float m2 = A.z + Bv.z + eav * Wv.z; m2 = m2 > 0.f ? m2 : 0.2f * m2;
        float m3 = A.w + Bv.w + eav * Wv.w; m3 = m3 > 0.f ? m3 : 0.2f * m3;
        lg += m0 * Tv.x + m1 * Tv.y + m2 * Tv.z + m3 * Tv.w;
    }
    logit[t] = lg;
    atomicMax(&mxenc[(size_t)dst * 8 + h], fenc(lg));
}

// ---------------------------------------------------------------------------
// Edge pass 2: a = exp(logit - mx[dst]); logit <- a; ssum[dst,h] += a
// ---------------------------------------------------------------------------
__global__ void gat_edge_exp(const int* __restrict__ ei, float* __restrict__ logit,
                             const unsigned* __restrict__ mxenc, float* __restrict__ ssum,
                             int E_) {
    int t = blockIdx.x * blockDim.x + threadIdx.x;
    if (t >= E_ * 8) return;
    int e = t >> 3, h = t & 7;
    int dst = ei[E_ + e];
    float a = __expf(logit[t] - fdec(mxenc[(size_t)dst * 8 + h]));
    logit[t] = a;
    atomicAdd(&ssum[(size_t)dst * 8 + h], a);
}

// ---------------------------------------------------------------------------
// Edge pass 3: gout[dst, h*16+c] += xl[src, h*16+c] * (a / ssum[dst,h])
// ---------------------------------------------------------------------------
__global__ void gat_edge_scatter(const float* __restrict__ xl, const int* __restrict__ ei,
                                 const float* __restrict__ logit, const float* __restrict__ ssum,
                                 float* __restrict__ gout, int E_) {
    int t = blockIdx.x * blockDim.x + threadIdx.x;
    if (t >= E_ * 8) return;
    int e = t >> 3, h = t & 7;
    int src = ei[e], dst = ei[E_ + e];
    float alpha = logit[t] / (ssum[(size_t)dst * 8 + h] + 1e-16f);
    const float* xs = xl + (size_t)src * HC + h * 16;
    float*       go = gout + (size_t)dst * HC + h * 16;
#pragma unroll
    for (int c = 0; c < 16; ++c) atomicAdd(&go[c], xs[c] * alpha);
}

// ---------------------------------------------------------------------------
// Init kernels
// ---------------------------------------------------------------------------
__global__ void init_maxsum(unsigned* __restrict__ mxenc, float* __restrict__ ssum, int n8) {
    int t = blockIdx.x * blockDim.x + threadIdx.x;
    if (t >= n8) return;
    mxenc[t] = 0u;       // encodes "below every float"
    ssum[t]  = 0.f;
}
__global__ void init_gout(float* __restrict__ gout, const float* __restrict__ bias, int n_) {
    int t = blockIdx.x * blockDim.x + threadIdx.x;
    if (t >= n_ * HC) return;
    gout[t] = bias[t & (HC - 1)];
}

// ---------------------------------------------------------------------------
// BatchNorm: stage 1 partial column sums (coalesced: 128 threads = 128 cols)
// ---------------------------------------------------------------------------
__global__ void bn_stats1(const float* __restrict__ gout, float* __restrict__ pS,
                          float* __restrict__ pS2, int n_) {
    int chunk = blockIdx.x, j = threadIdx.x;
    int r0 = chunk * STAT_ROWS;
    int r1 = r0 + STAT_ROWS; if (r1 > n_) r1 = n_;
    float s = 0.f, s2 = 0.f;
    for (int r = r0; r < r1; ++r) {
        float v = gout[(size_t)r * HC + j];
        s += v; s2 += v * v;
    }
    pS[chunk * HC + j] = s;
    pS2[chunk * HC + j] = s2;
}
__global__ void bn_stats2(const float* __restrict__ pS, const float* __restrict__ pS2,
                          const float* __restrict__ g, const float* __restrict__ bb,
                          float* __restrict__ scale, float* __restrict__ shift,
                          int nchunks, float invN) {
    int j = threadIdx.x;
    float s = 0.f, s2 = 0.f;
    for (int c = 0; c < nchunks; ++c) { s += pS[c * HC + j]; s2 += pS2[c * HC + j]; }
    float mean = s * invN;
    float var  = s2 * invN - mean * mean;
    float sc   = g[j] * rsqrtf(var + 1e-5f);
    scale[j] = sc;
    shift[j] = bb[j] - mean * sc;
}
__global__ void bn_apply(const float* __restrict__ gout, const float* __restrict__ scale,
                         const float* __restrict__ shift, const float* __restrict__ resid,
                         float* __restrict__ hout, int n_, int has_resid) {
    int t = blockIdx.x * blockDim.x + threadIdx.x;
    if (t >= n_ * HC) return;
    int j = t & (HC - 1);
    float v = gout[t] * scale[j] + shift[j];
    if (has_resid) v += resid[t];
    hout[t] = v > 0.f ? v : 0.f;
}

// ---------------------------------------------------------------------------
// Global mean pool: pooled[b,j] = mean over 360 nodes of h
// ---------------------------------------------------------------------------
__global__ void pool_mean(const float* __restrict__ h, float* __restrict__ pooled) {
    int b = blockIdx.x, j = threadIdx.x;   // 128 threads
    float s = 0.f;
    const float* base = h + (size_t)b * NODESPG * HC + j;
    for (int i = 0; i < NODESPG; ++i) s += base[(size_t)i * HC];
    pooled[b * HC + j] = s * (1.f / (float)NODESPG);
}

// ---------------------------------------------------------------------------
// Head: per-graph feature MLPs + 3-layer FC. One block (128 thr) per graph.
// ---------------------------------------------------------------------------
__global__ void head_kernel(const float* __restrict__ pd, const float* __restrict__ pf,
                            const float* __restrict__ gd, const float* __restrict__ gf,
                            const float* mW1, const float* mW2, const float* mb1, const float* mb2,
                            const float* bW1, const float* bW2, const float* bb1, const float* bb2,
                            const float* dW1, const float* dW2, const float* db1, const float* db2,
                            const float* fW1, const float* fW2, const float* fb1, const float* fb2,
                            const float* fc0W, const float* fc0b,
                            const float* fc1W, const float* fc1b,
                            const float* fc2W, const float* fc2b,
                            float* __restrict__ out) {
    __shared__ float z[368];
    __shared__ float a1[128];
    __shared__ float a2[64];
    __shared__ float t1[32];
    int b = blockIdx.x, tid = threadIdx.x;

    z[tid]       = pd[b * HC + tid];
    z[128 + tid] = pf[b * HC + tid];
    __syncthreads();

    // meta: 2 -> 16 -> 16
    if (tid < 16) {
        float g0 = gd[b * 13 + 0], g1 = gd[b * 13 + 1];
        float v = g0 * mW1[tid] + g1 * mW1[16 + tid] + mb1[tid];
        t1[tid] = v > 0.f ? v : 0.f;
    }
    __syncthreads();
    if (tid < 16) {
        float s = mb2[tid];
        for (int k = 0; k < 16; ++k) s += t1[k] * mW2[k * 16 + tid];
        z[256 + tid] = s > 0.f ? s : 0.f;
    }
    __syncthreads();
    // bio: 7 -> 32 -> 32
    if (tid < 32) {
        float s = bb1[tid];
        for (int k = 0; k < 7; ++k) s += gd[b * 13 + 2 + k] * bW1[k * 32 + tid];
        t1[tid] = s > 0.f ? s : 0.f;
    }
    __syncthreads();
    if (tid < 32) {
        float s = bb2[tid];
        for (int k = 0; k < 32; ++k) s += t1[k] * bW2[k * 32 + tid];
        z[272 + tid] = s > 0.f ? s : 0.f;
    }
    __syncthreads();
    // gdti: 4 -> 32 -> 32
    if (tid < 32) {
        float s = db1[tid];
        for (int k = 0; k < 4; ++k) s += gd[b * 13 + 9 + k] * dW1[k * 32 + tid];
        t1[tid] = s > 0.f ? s : 0.f;
    }
    __syncthreads();
    if (tid < 32) {
        float s = db2[tid];
        for (int k = 0; k < 32; ++k) s += t1[k] * dW2[k * 32 + tid];
        z[304 + tid] = s > 0.f ? s : 0.f;
    }
    __syncthreads();
    // gfmri: 4 -> 32 -> 32
    if (tid < 32) {
        float s = fb1[tid];
        for (int k = 0; k < 4; ++k) s += gf[b * 4 + k] * fW1[k * 32 + tid];
        t1[tid] = s > 0.f ? s : 0.f;
    }
    __syncthreads();
    if (tid < 32) {
        float s = fb2[tid];
        for (int k = 0; k < 32; ++k) s += t1[k] * fW2[k * 32 + tid];
        z[336 + tid] = s > 0.f ? s : 0.f;
    }
    __syncthreads();
    // fc1: 368 -> 128 (relu)
    {
        float s = fc0b[tid];
        for (int k = 0; k < 368; ++k) s += z[k] * fc0W[k * 128 + tid];
        a1[tid] = s > 0.f ? s : 0.f;
    }
    __syncthreads();
    // fc2: 128 -> 64 (relu)
    if (tid < 64) {
        float s = fc1b[tid];
        for (int k = 0; k < 128; ++k) s += a1[k] * fc1W[k * 64 + tid];
        a2[tid] = s > 0.f ? s : 0.f;
    }
    __syncthreads();
    // fc3: 64 -> 1
    if (tid == 0) {
        float s = fc2b[0];
        for (int k = 0; k < 64; ++k) s += a2[k] * fc2W[k];
        out[b] = s;
    }
}

// ---------------------------------------------------------------------------
// Host orchestration.
// params flattened jax-style (dicts sorted by key, lists in order):
//   0..3   bio {W1,W2,b1,b2}
//   4..11  dti_bn  4 x {b,g}
//   12..39 dti_gat 4 x {We,Wl,Wr,att,bias,bl,br}
//   40,41  embed_W, embed_b
//   42..47 fc 3 x {W,b}
//   48..55 fmri_bn
//   56..83 fmri_gat
//   84..87 gdti {W1,W2,b1,b2}
//   88..91 gfmri
//   92..95 meta
//   96 x_dti, 97 x_fmri, 98 ei_dti, 99 ea_dti, 100 ei_fmri, 101 ea_fmri,
//   102 batch, 103 gfeat_dti, 104 gfeat_fmri
// ---------------------------------------------------------------------------
extern "C" void kernel_launch(void* const* d_in, const int* in_sizes, int n_in,
                              void* d_out, int out_size, void* d_ws, size_t ws_size,
                              hipStream_t stream) {
    const int N_ = NNODES, E_ = NEDGES, B_ = NGRAPH;
    auto F = [&](int i) { return (const float*)d_in[i]; };
    auto I = [&](int i) { return (const int*)d_in[i]; };

    float* ws = (float*)d_ws;
    size_t off = 0;
    float* bufA  = ws + off; off += (size_t)N_ * HC;
    float* bufB  = ws + off; off += (size_t)N_ * HC;
    float* xl    = ws + off; off += (size_t)N_ * HC;
    float* xr    = ws + off; off += (size_t)N_ * HC;  // also the scatter target (gout)
    float* alpha = ws + off; off += (size_t)E_ * 8;
    unsigned* mxenc = (unsigned*)(ws + off); off += (size_t)N_ * 8;
    float* ssum  = ws + off; off += (size_t)N_ * 8;
    float* pS    = ws + off; off += STAT_CHUNKS * HC;
    float* pS2   = ws + off; off += STAT_CHUNKS * HC;
    float* scale = ws + off; off += HC;
    float* shift = ws + off; off += HC;
    float* pd    = ws + off; off += (size_t)B_ * HC;
    float* pf    = ws + off; off += (size_t)B_ * HC;
    // packed bf16 weight fragments (32B-aligned: off stays a multiple of 8)
    unsigned short* plbuf = (unsigned short*)(ws + off); off += 4 * 8 * 32 * 16 / 2;
    unsigned short* prbuf = (unsigned short*)(ws + off); off += 4 * 8 * 32 * 16 / 2;
    (void)ws_size; (void)n_in; (void)in_sizes; (void)out_size;

    const int TB = 256;
    struct ModDesc { int gat_base, bn_base, x_idx, ei_idx, ea_idx; float* pooled; };
    ModDesc mods[2] = { {12, 4, 96, 98, 99, pd}, {56, 48, 97, 100, 101, pf} };
    const int dins[4] = {64, HC, HC, HC};

    for (int mi = 0; mi < 2; ++mi) {
        const ModDesc& md = mods[mi];
        const int* ei = I(md.ei_idx);
        const float* ea = F(md.ea_idx);

        // node embedding -> bufA as [N,64]
        {
            int nt = N_ * 64;
            embed_kernel<<<(nt + TB - 1) / TB, TB, 0, stream>>>(
                F(md.x_idx), F(40), F(41), bufA, N_);
        }

        float* cur = bufA;
        float* nxt = bufB;
        for (int L = 0; L < 4; ++L) {
            int gb = md.gat_base + L * 7;   // We,Wl,Wr,att,bias,bl,br
            int bb = md.bn_base + L * 2;    // b,g
            int din = dins[L];
            int ksteps = din / 32;

            // pre-pack both weight matrices into bf16 WMMA fragment order
            {
                int nt = 2 * ksteps * 8 * 32;
                pack_wmma_b<<<(nt + TB - 1) / TB, TB, 0, stream>>>(
                    F(gb + 1), F(gb + 2), plbuf, prbuf, ksteps);
            }
            // xl / xr GEMMs (WMMA bf16, one wave per 16-row strip)
            {
                int waves = 2 * (N_ / 16);
                int nt = waves * 32;
                if (din == 64)
                    gemm_strip_wmma<64><<<(nt + TB - 1) / TB, TB, 0, stream>>>(
                        cur, plbuf, prbuf, F(gb + 5), F(gb + 6), xl, xr, N_);
                else
                    gemm_strip_wmma<128><<<(nt + TB - 1) / TB, TB, 0, stream>>>(
                        cur, plbuf, prbuf, F(gb + 5), F(gb + 6), xl, xr, N_);
            }
            // softmax stats init
            {
                int nt = N_ * 8;
                init_maxsum<<<(nt + TB - 1) / TB, TB, 0, stream>>>(mxenc, ssum, nt);
            }
            // edge pass 1: logits + segment max
            {
                int nt = E_ * 8;
                gat_edge_logits<<<(nt + TB - 1) / TB, TB, 0, stream>>>(
                    xl, xr, ei, ea, F(gb + 0), F(gb + 3), alpha, mxenc, E_);
            }
            // xr is now dead -> reuse as scatter target, seeded with GAT bias
            {
                int nt = N_ * HC;
                init_gout<<<(nt + TB - 1) / TB, TB, 0, stream>>>(xr, F(gb + 4), N_);
            }
            // edge pass 2: exp + segment sum
            {
                int nt = E_ * 8;
                gat_edge_exp<<<(nt + TB - 1) / TB, TB, 0, stream>>>(
                    ei, alpha, mxenc, ssum, E_);
            }
            // edge pass 3: weighted scatter into xr(=gout)
            {
                int nt = E_ * 8;
                gat_edge_scatter<<<(nt + TB - 1) / TB, TB, 0, stream>>>(
                    xl, ei, alpha, ssum, xr, E_);
            }
            // batch norm stats + apply (+relu, +residual for L>=1)
            bn_stats1<<<STAT_CHUNKS, HC, 0, stream>>>(xr, pS, pS2, N_);
            bn_stats2<<<1, HC, 0, stream>>>(pS, pS2, F(bb + 1), F(bb + 0),
                                            scale, shift, STAT_CHUNKS, 1.f / (float)N_);
            {
                int nt = N_ * HC;
                bn_apply<<<(nt + TB - 1) / TB, TB, 0, stream>>>(
                    xr, scale, shift, cur, nxt, N_, L >= 1 ? 1 : 0);
            }
            float* tmp = cur; cur = nxt; nxt = tmp;
        }
        pool_mean<<<B_, HC, 0, stream>>>(cur, md.pooled);
    }

    head_kernel<<<B_, HC, 0, stream>>>(
        pd, pf, F(103), F(104),
        F(92), F(93), F(94), F(95),      // meta
        F(0),  F(1),  F(2),  F(3),       // bio
        F(84), F(85), F(86), F(87),      // gdti
        F(88), F(89), F(90), F(91),      // gfmri
        F(42), F(43), F(44), F(45), F(46), F(47),
        (float*)d_out);
}
